// MQCA_70866960384415
// MI455X (gfx1250) — matrix operand order repaired and checked by reading, not compiled, not run on hardware
//
#include <hip/hip_runtime.h>
#include <hip/hip_bf16.h>
#include <math.h>

typedef _Float16 h16;
typedef __attribute__((ext_vector_type(16))) _Float16 v16h;
typedef __attribute__((ext_vector_type(8)))  _Float16 v8h;
typedef __attribute__((ext_vector_type(4)))  _Float16 v4h;
typedef __attribute__((ext_vector_type(8)))  float    v8f;

// ---- optional CDNA5 async global->LDS path (guarded; falls back cleanly) ----
#if __has_builtin(__builtin_amdgcn_global_load_async_to_lds_b128) && \
    __has_builtin(__builtin_amdgcn_s_wait_asynccnt)
#define HAVE_ASYNC_LDS 1
typedef __attribute__((ext_vector_type(4))) int i4;
typedef __attribute__((address_space(1))) i4 as1_i4;   // printed as "int4 __device__*"
typedef __attribute__((address_space(3))) i4 as3_i4;   // LDS
__device__ __forceinline__ as1_i4* to_global_as(const void* p)
{
    // flat global address == AS1 address numerically
    return (as1_i4*)(unsigned long long)(size_t)p;
}
__device__ __forceinline__ as3_i4* to_lds_as(const void* p)
{
    // ISA aperture rule: LDS offset = flat_addr[31:0]
    return (as3_i4*)(unsigned int)(size_t)p;
}
#else
#define HAVE_ASYNC_LDS 0
#endif

// ---------------------------------------------------------------------------
// f32 -> f16 conversion (vectorized), n multiple of 4
// ---------------------------------------------------------------------------
__global__ void f32tof16_k(const float* __restrict__ in, h16* __restrict__ out, size_t n)
{
    size_t i = ((size_t)blockIdx.x * 256 + threadIdx.x) * 4;
    if (i + 3 < n) {
        float4 v = *(const float4*)(in + i);
        v4h o; o[0] = (h16)v.x; o[1] = (h16)v.y; o[2] = (h16)v.z; o[3] = (h16)v.w;
        *(v4h*)(out + i) = o;
    }
}

// ---------------------------------------------------------------------------
// WMMA GEMM (f16 in, f32 out):  Out[b][o][m] = sum_c W[o][c] * In[b][c][m]
// 256 threads = 8 wave32; block tile 32(O) x 128(M); K step 32.
// Double-buffered transposed B tile in LDS; per-wave 16x32 (2 accumulators,
// shared A fragment). A fragments load straight from L2-resident weights.
// ---------------------------------------------------------------------------
__global__ __launch_bounds__(256)
void gemm_wmma_k(const h16* __restrict__ Wt, const h16* __restrict__ In,
                 float* __restrict__ Out, int O, int Cc, int M)
{
    const int b = blockIdx.z;
    In  += (size_t)b * Cc * M;
    Out += (size_t)b * O  * M;
    const int m0   = blockIdx.x * 128;
    const int o0   = blockIdx.y * 32;
    const int tid  = threadIdx.x;
    const int lane = tid & 31;
    const int wv   = tid >> 5;         // 0..7
    const int wo   = (wv >> 2) * 16;   // O sub-tile: 0 or 16
    const int wm   = (wv & 3)  * 32;   // M sub-tile: 0,32,64,96

    __shared__ h16 lBt[2][128 * 40];   // transposed B: [m][k], 80B row stride

    v8f acc0 = {}, acc1 = {};

    const int lm  = lane & 15;
    const int kbA = (lane & 16) ? 8  : 0;
    const int kbB = (lane & 16) ? 8  : 0;

    const int sr   = tid >> 3;         // staging k-row 0..31
    const int sseg = (tid & 7) * 16;   // staging m-seg 0..112

    const h16* arow = Wt + (size_t)(o0 + wo + lm) * Cc;
    const h16* stgp = In + (size_t)sr * M + (m0 + sseg);

    // prologue: stage buffer 0 (K rows 0..31)
    {
        v8h d0 = *(const v8h*)(stgp);
        v8h d1 = *(const v8h*)(stgp + 8);
        #pragma unroll
        for (int j = 0; j < 8; ++j) {
            lBt[0][(sseg + j)     * 40 + sr] = d0[j];
            lBt[0][(sseg + 8 + j) * 40 + sr] = d1[j];
        }
    }
    __syncthreads();

    const int nsteps = Cc >> 5;
    for (int s = 0; s < nsteps; ++s) {
        const int cur = s & 1;
        const int c0  = s << 5;
        const bool more = (s + 1 < nsteps);

        // issue next-slab global loads first (latency hidden under WMMA)
        v8h d0, d1;
        if (more) {
            const h16* np = stgp + (size_t)(c0 + 32) * M;
            d0 = *(const v8h*)(np);
            d1 = *(const v8h*)(np + 8);
        }
        if (s + 2 < nsteps)   // global_prefetch_b8 of the slab after next
            __builtin_prefetch(stgp + (size_t)(c0 + 64) * M, 0, 1);

        // A fragment: 2 x global_load_b128 (K-contiguous weights)
        v8h alo = *(const v8h*)&arow[c0 + kbA];
        v8h ahi = *(const v8h*)&arow[c0 + kbA + 16];
        v16h a  = __builtin_shufflevector(alo, ahi,
                    0,1,2,3,4,5,6,7,8,9,10,11,12,13,14,15);

        // B fragments: 2 x (2 x ds_load_b128), contiguous per lane
        const h16* br0 = &lBt[cur][(wm + lm)      * 40 + kbB];
        const h16* br1 = &lBt[cur][(wm + 16 + lm) * 40 + kbB];
        v8h b0lo = *(const v8h*)&br0[0], b0hi = *(const v8h*)&br0[16];
        v8h b1lo = *(const v8h*)&br1[0], b1hi = *(const v8h*)&br1[16];
        v16h bf0 = __builtin_shufflevector(b0lo, b0hi,
                    0,1,2,3,4,5,6,7,8,9,10,11,12,13,14,15);
        v16h bf1 = __builtin_shufflevector(b1lo, b1hi,
                    0,1,2,3,4,5,6,7,8,9,10,11,12,13,14,15);

        acc0 = __builtin_amdgcn_wmma_f32_16x16x32_f16(false, a, false, bf0,
                                                      (short)0, acc0, false, false);
        acc1 = __builtin_amdgcn_wmma_f32_16x16x32_f16(false, a, false, bf1,
                                                      (short)0, acc1, false, false);

        // drain the staged loads into the other buffer
        if (more) {
            const int nxt = cur ^ 1;
            #pragma unroll
            for (int j = 0; j < 8; ++j) {
                lBt[nxt][(sseg + j)     * 40 + sr] = d0[j];
                lBt[nxt][(sseg + 8 + j) * 40 + sr] = d1[j];
            }
        }
        __syncthreads();
    }

    // D layout: lane = N (mod 16), VGPR v = M row (+8 for lanes 16-31)
    const int mb = (lane & 16) ? 8 : 0;
    float* op = Out + (size_t)(o0 + wo + mb) * M + m0;
    #pragma unroll
    for (int v = 0; v < 8; ++v) {
        op[(size_t)v * M + (wm + lm)]      = acc0[v];
        op[(size_t)v * M + (wm + 16 + lm)] = acc1[v];
    }
}

// ---------------------------------------------------------------------------
// Depthwise 3x3 (pad 1), NCHW, H=W=128
// ---------------------------------------------------------------------------
__global__ void dwconv3x3_k(const float* __restrict__ in, const float* __restrict__ w,
                            float* __restrict__ out, int CH)
{
    size_t idx = (size_t)blockIdx.x * blockDim.x + threadIdx.x;
    size_t total = (size_t)4 * CH * 128 * 128;
    if (idx >= total) return;
    int xc = (int)(idx & 127);
    int yc = (int)((idx >> 7) & 127);
    int ch = (int)((idx >> 14) % CH);
    const float* ip = in + (idx - (size_t)(yc * 128 + xc));  // plane base
    const float* wp = w + ch * 9;
    float acc = 0.f;
    #pragma unroll
    for (int dy = -1; dy <= 1; ++dy) {
        int yy = yc + dy;
        if (yy < 0 || yy > 127) continue;
        #pragma unroll
        for (int dx = -1; dx <= 1; ++dx) {
            int xx = xc + dx;
            if (xx < 0 || xx > 127) continue;
            acc += wp[(dy + 1) * 3 + (dx + 1)] * ip[yy * 128 + xx];
        }
    }
    out[idx] = acc;
}

// ---------------------------------------------------------------------------
// L2 norm over S=16384 per (b, channel)
// ---------------------------------------------------------------------------
__global__ void colnorm_k(const float* __restrict__ t, int CHtot, int choff,
                          float* __restrict__ nrm)
{
    int c = blockIdx.x, b = blockIdx.y, tid = threadIdx.x;
    const float* p = t + ((size_t)b * CHtot + choff + c) * 16384;
    float acc = 0.f;
    for (int l = tid; l < 16384; l += 256) { float v = p[l]; acc += v * v; }
    __shared__ float sm[256];
    sm[tid] = acc; __syncthreads();
    for (int s = 128; s > 0; s >>= 1) { if (tid < s) sm[tid] += sm[tid + s]; __syncthreads(); }
    if (tid == 0) nrm[b * 192 + c] = sqrtf(sm[0]);
}

// ---------------------------------------------------------------------------
// Full 24x24 Gram matrix per (b, head), single-pass HBM reads.
// Tile staging uses async global->LDS (b128) when the toolchain has it.
// ---------------------------------------------------------------------------
__global__ __launch_bounds__(256)
void attn_gram_k(const float* __restrict__ q, int qCH, int qOff,
                 const float* __restrict__ k, int kCH, int kOff,
                 const float* __restrict__ nq, const float* __restrict__ nk,
                 const float* __restrict__ temp, float* __restrict__ attn)
{
    int n = blockIdx.x, b = blockIdx.y, tid = threadIdx.x;
    const float* qb = q + ((size_t)b * qCH + qOff + n * 24) * 16384;
    const float* kb = k + ((size_t)b * kCH + kOff + n * 24) * 16384;
    __shared__ float qt[24 * 260];   // stride 260: keeps float4 16B-aligned
    __shared__ float kt[24 * 260];

    const int p0 = tid, p1 = tid + 256, p2 = tid + 512;
    const int d0 = p0 / 24, e0 = p0 % 24;
    const int d1 = p1 / 24, e1 = p1 % 24;
    const int p2c = (p2 < 576) ? p2 : 575;
    const int d2 = p2c / 24, e2 = p2c % 24;
    float a0 = 0.f, a1 = 0.f, a2 = 0.f;

    for (int l0 = 0; l0 < 16384; l0 += 256) {
        __syncthreads();
        #pragma unroll
        for (int it = 0; it < 6; ++it) {           // 24 rows x 64 float4
            int i = it * 256 + tid;                // 0..1535
            int r = i >> 6, c4 = (i & 63) * 4;
#if HAVE_ASYNC_LDS
            __builtin_amdgcn_global_load_async_to_lds_b128(
                to_global_as(&qb[(size_t)r * 16384 + l0 + c4]),
                to_lds_as(&qt[r * 260 + c4]), 0, 0);
            __builtin_amdgcn_global_load_async_to_lds_b128(
                to_global_as(&kb[(size_t)r * 16384 + l0 + c4]),
                to_lds_as(&kt[r * 260 + c4]), 0, 0);
#else
            *(float4*)&qt[r * 260 + c4] = *(const float4*)&qb[(size_t)r * 16384 + l0 + c4];
            *(float4*)&kt[r * 260 + c4] = *(const float4*)&kb[(size_t)r * 16384 + l0 + c4];
#endif
        }
#if HAVE_ASYNC_LDS
        __builtin_amdgcn_s_wait_asynccnt(0);
#endif
        __syncthreads();
        #pragma unroll 2
        for (int j = 0; j < 256; j += 4) {
            float4 xq, xk;
            xq = *(const float4*)&qt[d0 * 260 + j]; xk = *(const float4*)&kt[e0 * 260 + j];
            a0 += xq.x * xk.x + xq.y * xk.y + xq.z * xk.z + xq.w * xk.w;
            xq = *(const float4*)&qt[d1 * 260 + j]; xk = *(const float4*)&kt[e1 * 260 + j];
            a1 += xq.x * xk.x + xq.y * xk.y + xq.z * xk.z + xq.w * xk.w;
            xq = *(const float4*)&qt[d2 * 260 + j]; xk = *(const float4*)&kt[e2 * 260 + j];
            a2 += xq.x * xk.x + xq.y * xk.y + xq.z * xk.z + xq.w * xk.w;
        }
    }

    const float* nqb = nq + b * 192 + n * 24;
    const float* nkb = nk + b * 192 + n * 24;
    float* ab = attn + (size_t)(b * 8 + n) * 576;
    float tp = temp[n];
    ab[p0] = a0 / (fmaxf(nqb[d0], 1e-12f) * fmaxf(nkb[e0], 1e-12f)) * tp;
    ab[p1] = a1 / (fmaxf(nqb[d1], 1e-12f) * fmaxf(nkb[e1], 1e-12f)) * tp;
    if (p2 < 576)
        ab[p2] = a2 / (fmaxf(nqb[d2], 1e-12f) * fmaxf(nkb[e2], 1e-12f)) * tp;
}

__global__ void softmax24_k(float* attn)
{
    int d = blockIdx.x, n = blockIdx.y, b = blockIdx.z;
    float* row = attn + (((size_t)b * 8 + n) * 24 + d) * 24;
    if (threadIdx.x == 0) {
        float mx = row[0];
        for (int e = 1; e < 24; ++e) mx = fmaxf(mx, row[e]);
        float s = 0.f, tmp[24];
        for (int e = 0; e < 24; ++e) { tmp[e] = expf(row[e] - mx); s += tmp[e]; }
        float inv = 1.f / s;
        for (int e = 0; e < 24; ++e) row[e] = tmp[e] * inv;
    }
}

// out[b][n*24+d][l] = sum_e attn[d][e] * v[b][...][l]
__global__ void attn_apply_k(const float* __restrict__ attn,
                             const float* __restrict__ v, int vCH, int vOff,
                             float* __restrict__ out)
{
    int n = blockIdx.y, b = blockIdx.z;
    int l = blockIdx.x * 256 + threadIdx.x;
    __shared__ float at[576];
    for (int i = threadIdx.x; i < 576; i += 256)
        at[i] = attn[((size_t)b * 8 + n) * 576 + i];
    __syncthreads();
    float ve[24];
    #pragma unroll
    for (int e = 0; e < 24; ++e)
        ve[e] = v[((size_t)b * vCH + vOff + n * 24 + e) * 16384 + l];
    #pragma unroll
    for (int d = 0; d < 24; ++d) {
        float s = 0.f;
        #pragma unroll
        for (int e = 0; e < 24; ++e) s += at[d * 24 + e] * ve[e];
        out[((size_t)b * 192 + n * 24 + d) * 16384 + l] = s;
    }
}

// ---------------------------------------------------------------------------
// 128-point radix-2 DIT FFT in LDS (bit-reversed load -> natural order out)
// ---------------------------------------------------------------------------
__device__ inline void fft128_sh(float* re, float* im, int t, float sign)
{
    #pragma unroll
    for (int s = 1; s <= 7; ++s) {
        int mh = 1 << (s - 1);
        if (t < 64) {
            int pos = t & (mh - 1);
            int grp = t >> (s - 1);
            int i = (grp << s) + pos;
            int j = i + mh;
            float ang = sign * 3.14159265358979323846f * (float)pos / (float)mh;
            float cr = __cosf(ang), sr = __sinf(ang);
            float tr = re[j] * cr - im[j] * sr;
            float ti = re[j] * sr + im[j] * cr;
            re[j] = re[i] - tr; im[j] = im[i] - ti;
            re[i] += tr;        im[i] += ti;
        }
        __syncthreads();
    }
}

// rows: real 128 -> 65 complex; f layout [b][2C][128][65] (real: c, imag: c+C)
__global__ void fft_row_fwd_k(const float* __restrict__ x, float* __restrict__ f, int C)
{
    int row = blockIdx.x, c = blockIdx.y, b = blockIdx.z, t = threadIdx.x;
    __shared__ float re[128], im[128];
    int rv = (int)(__brev((unsigned)t) >> 25);
    re[rv] = x[(((size_t)b * C + c) * 128 + row) * 128 + t];
    im[rv] = 0.f;
    __syncthreads();
    fft128_sh(re, im, t, -1.f);
    if (t < 65) {
        size_t o = (((size_t)b * 2 * C + c) * 128 + row) * 65 + t;
        f[o] = re[t];
        f[o + (size_t)C * 128 * 65] = im[t];
    }
}

// columns: in-place complex FFT over h for each of 65 columns
__global__ void fft_col_k(float* __restrict__ f, int C, float sign, float scale)
{
    int col = blockIdx.x, c = blockIdx.y, b = blockIdx.z, t = threadIdx.x;
    float* fre = f + ((size_t)b * 2 * C + c) * 128 * 65 + col;
    float* fim = fre + (size_t)C * 128 * 65;
    __shared__ float re[128], im[128];
    int rv = (int)(__brev((unsigned)t) >> 25);
    re[rv] = fre[(size_t)t * 65];
    im[rv] = fim[(size_t)t * 65];
    __syncthreads();
    fft128_sh(re, im, t, sign);
    fre[(size_t)t * 65] = re[t] * scale;
    fim[(size_t)t * 65] = im[t] * scale;
}

// rows inverse: 65 complex (+ Hermitian extension) -> 128 real
__global__ void fft_row_inv_k(const float* __restrict__ f, float* __restrict__ out, int C)
{
    int row = blockIdx.x, c = blockIdx.y, b = blockIdx.z, t = threadIdx.x;
    const float* fre = f + (((size_t)b * 2 * C + c) * 128 + row) * 65;
    const float* fim = fre + (size_t)C * 128 * 65;
    __shared__ float re[128], im[128];
    int rv = (int)(__brev((unsigned)t) >> 25);
    float xr, xi;
    if (t < 65) { xr = fre[t];        xi = fim[t]; }
    else        { xr = fre[128 - t];  xi = -fim[128 - t]; }
    re[rv] = xr; im[rv] = xi;
    __syncthreads();
    fft128_sh(re, im, t, 1.f);
    out[(((size_t)b * C + c) * 128 + row) * 128 + t] = re[t] * (1.f / 128.f);
}

__global__ void gelu_k(float* __restrict__ p, size_t n)
{
    size_t i = (size_t)blockIdx.x * 256 + threadIdx.x;
    if (i < n) {
        float x = p[i];
        p[i] = 0.5f * x * (1.f + erff(x * 0.70710678118654752f));
    }
}

// ---------------------------------------------------------------------------
static void cvt16(const float* src, h16* dst, size_t n, hipStream_t s)
{
    f32tof16_k<<<(unsigned)((n / 4 + 255) / 256), 256, 0, s>>>(src, dst, n);
}

static void conv1x1(const float* W, const float* In, float* Out,
                    int O, int Cc, int M, h16* wh, h16* inh, hipStream_t s)
{
    cvt16(W, wh, (size_t)O * Cc, s);
    cvt16(In, inh, 4ull * Cc * M, s);
    gemm_wmma_k<<<dim3((unsigned)(M / 128), (unsigned)(O / 32), 4), 256, 0, s>>>(
        wh, inh, Out, O, Cc, M);
}

static void run_attn(const float* q, int qCH, int qOff,
                     const float* k, int kCH, int kOff,
                     const float* v, int vCH, int vOff,
                     const float* temp, float* nq, float* nk, float* attn,
                     float* out, hipStream_t stream)
{
    colnorm_k<<<dim3(192, 4), 256, 0, stream>>>(q, qCH, qOff, nq);
    colnorm_k<<<dim3(192, 4), 256, 0, stream>>>(k, kCH, kOff, nk);
    attn_gram_k<<<dim3(8, 4), 256, 0, stream>>>(q, qCH, qOff, k, kCH, kOff,
                                                nq, nk, temp, attn);
    softmax24_k<<<dim3(24, 8, 4), 32, 0, stream>>>(attn);
    attn_apply_k<<<dim3(64, 8, 4), 256, 0, stream>>>(attn, v, vCH, vOff, out);
}

extern "C" void kernel_launch(void* const* d_in, const int* in_sizes, int n_in,
                              void* d_out, int out_size, void* d_ws, size_t ws_size,
                              hipStream_t stream)
{
    (void)in_sizes; (void)n_in; (void)out_size; (void)ws_size;
    const float* x          = (const float*)d_in[0];
    const float* w_qkv1     = (const float*)d_in[1];
    const float* w_qkv1_dw  = (const float*)d_in[2];
    const float* w_proj1    = (const float*)d_in[3];
    const float* w_fdfp1    = (const float*)d_in[4];
    const float* w_fdfp2    = (const float*)d_in[5];
    const float* w_kv2      = (const float*)d_in[6];
    const float* w_kv2_dw   = (const float*)d_in[7];
    const float* w_proj2    = (const float*)d_in[8];
    const float* temp       = (const float*)d_in[9];

    const size_t S = 16384;      // 128*128
    const size_t F = 8320;       // 128*65

    float* ws = (float*)d_ws;
    size_t off = 0;
    float* t0    = ws + off; off += 4ull * 576 * S;  // qkv1 raw / kv2 raw
    float* t1    = ws + off; off += 4ull * 576 * S;  // qkv1 dw  / kv2 dw
    float* out1  = ws + off; off += 4ull * 192 * S;  // attn out (stage 1 & 2)
    float* out1p = ws + off; off += 4ull * 192 * S;  // proj1 out
    float* fA    = ws + off; off += 4ull * 384 * F;  // freq buf A
    float* fB    = ws + off; off += 4ull * 384 * F;  // freq buf B
    float* q2    = ws + off; off += 4ull * 192 * S;  // irfft2 output
    float* nq    = ws + off; off += 4ull * 192;
    float* nk    = ws + off; off += 4ull * 192;
    float* attn  = ws + off; off += 4ull * 8 * 24 * 24;
    h16*  wh     = (h16*)(ws + off); off += (576ull * 192) / 2;       // f16 weights
    h16*  inh    = (h16*)(ws + off); off += (4ull * 384 * F) / 2;     // f16 activations

    // ---- Stage 1: qkv -> dwconv -> channel attention -> proj ----
    conv1x1(w_qkv1, x, t0, 576, 192, (int)S, wh, inh, stream);
    dwconv3x3_k<<<(unsigned)((4ull * 576 * S) / 256), 256, 0, stream>>>(
        t0, w_qkv1_dw, t1, 576);
    run_attn(t1, 576, 0, t1, 576, 192, t1, 576, 384, temp, nq, nk, attn, out1, stream);
    conv1x1(w_proj1, out1, out1p, 192, 192, (int)S, wh, inh, stream);

    // ---- FDFP: rfft2 -> conv1x1 -> gelu -> conv1x1 -> irfft2 ----
    fft_row_fwd_k<<<dim3(128, 192, 4), 128, 0, stream>>>(x, fA, 192);
    fft_col_k<<<dim3(65, 192, 4), 128, 0, stream>>>(fA, 192, -1.f, 1.f / 128.f);
    conv1x1(w_fdfp1, fA, fB, 384, 384, (int)F, wh, inh, stream);
    gelu_k<<<(unsigned)((4ull * 384 * F + 255) / 256), 256, 0, stream>>>(fB, 4ull * 384 * F);
    conv1x1(w_fdfp2, fB, fA, 384, 384, (int)F, wh, inh, stream);
    fft_col_k<<<dim3(65, 192, 4), 128, 0, stream>>>(fA, 192, 1.f, 1.f);
    fft_row_inv_k<<<dim3(128, 192, 4), 128, 0, stream>>>(fA, q2, 192);

    // ---- Stage 2: kv from stage-1 out, q from frequency branch ----
    conv1x1(w_kv2, out1p, t0, 384, 192, (int)S, wh, inh, stream);
    dwconv3x3_k<<<(unsigned)((4ull * 384 * S) / 256), 256, 0, stream>>>(
        t0, w_kv2_dw, t1, 384);
    run_attn(q2, 192, 0, t1, 384, 0, t1, 384, 192, temp, nq, nk, attn, out1, stream);
    conv1x1(w_proj2, out1, (float*)d_out, 192, 192, (int)S, wh, inh, stream);
}
